// DKVMN_c_q_18829136626098
// MI455X (gfx1250) — matrix operand order repaired and checked by reading, not compiled, not run on hardware
//
#include <hip/hip_runtime.h>
#include <hip/hip_bf16.h>

typedef __attribute__((ext_vector_type(16))) _Float16 v16h;
typedef __attribute__((ext_vector_type(8)))  float    v8f;

#define NUM_Q  1000
#define DIM_S  128
#define SIZE_M 50
#define BB     64
#define TT     200
#define BT     (BB * TT)          // 12800 positions
#define TILES  (BT / 16)          // 800 16-row tiles
#define WPB    4                  // waves per block
#define NBLK1  (TILES / WPB)      // 200 blocks for GEMM phases

// ---------------------------------------------------------------------------
// vectorized helpers (all call sites fully unrolled -> arrays stay in VGPRs)
// ---------------------------------------------------------------------------
__device__ __forceinline__ void load8a(const float* __restrict__ p, float* d) {
    const float4 a = ((const float4*)p)[0];
    const float4 b = ((const float4*)p)[1];
    d[0] = a.x; d[1] = a.y; d[2] = a.z; d[3] = a.w;
    d[4] = b.x; d[5] = b.y; d[6] = b.z; d[7] = b.w;
}

__device__ __forceinline__ void store8a(float* __restrict__ p, const float* d) {
    float4 a, b;
    a.x = d[0]; a.y = d[1]; a.z = d[2]; a.w = d[3];
    b.x = d[4]; b.y = d[5]; b.z = d[6]; b.w = d[7];
    ((float4*)p)[0] = a;
    ((float4*)p)[1] = b;
}

// B fragment: f32 p[0..7] and p[16..23] -> 16 f16 halves (optionally scaled by mask)
__device__ __forceinline__ v16h loadB16(const float* __restrict__ p, float s) {
    v16h o;
    #pragma unroll
    for (int i = 0; i < 4; ++i) {
        const float4 f = ((const float4*)p)[i + (i >> 1) * 2];
        o[4 * i + 0] = (_Float16)(f.x * s);
        o[4 * i + 1] = (_Float16)(f.y * s);
        o[4 * i + 2] = (_Float16)(f.z * s);
        o[4 * i + 3] = (_Float16)(f.w * s);
    }
    return o;
}

// ---------------------------------------------------------------------------
// Phase 1: gather k/v, WMMA GEMMs for w-logits / e / a, softmax(w)
// one wave handles 16 consecutive (b,t) positions
// A-frag 16x32 f16 layout: halves 0..7 <- K = c*32 + g*8 + h (contiguous 8),
//                          halves 8..15 <- K = c*32 + 16 + g*8 + (h-8)
// ---------------------------------------------------------------------------
__global__ void __launch_bounds__(128)
dkvmn_phase1(const int* __restrict__ q, const int* __restrict__ r,
             const int* __restrict__ pid,
             const float* __restrict__ pid_emb, const float* __restrict__ k_emb,
             const float* __restrict__ v_emb,   const float* __restrict__ Mk,
             const float* __restrict__ e_W, const float* __restrict__ e_b,
             const float* __restrict__ a_W, const float* __restrict__ a_b,
             float* __restrict__ ws_w, float* __restrict__ ws_e,
             float* __restrict__ ws_a, float* __restrict__ ws_k)
{
    __shared__ float lds_logits[WPB][16][64];

    const int wid  = threadIdx.x >> 5;
    const int lane = threadIdx.x & 31;
    const int g    = lane >> 4;       // lane group
    const int mrow = lane & 15;       // A-row / B-column / D-column index
    const int tile = blockIdx.x * WPB + wid;
    const int p0   = tile * 16;
    const int pos  = p0 + mrow;

    const int qi = q[pos];
    const int ri = r[pos];
    const int pi = pid[pos];
    const int xi = qi + NUM_Q * ri;
    const float* pe = pid_emb + (size_t)pi * DIM_S;
    const float* ke = k_emb   + (size_t)qi * DIM_S;
    const float* ve = v_emb   + (size_t)xi * DIM_S;
    float*       kw = ws_k    + (size_t)pos * DIM_S;

    // Build A fragments for k and v directly in WMMA layout; spill k (f32) to ws.
    v16h ka[4], va[4];
    #pragma unroll
    for (int c = 0; c < 4; ++c) {
        const int k0 = c * 32 + g * 8;        // halves 0..7
        const int k1 = c * 32 + 16 + g * 8;   // halves 8..15
        float peb[16], keb[16], veb[16], ksp[16];
        load8a(pe + k0, peb); load8a(pe + k1, peb + 8);
        load8a(ke + k0, keb); load8a(ke + k1, keb + 8);
        load8a(ve + k0, veb); load8a(ve + k1, veb + 8);
        #pragma unroll
        for (int h = 0; h < 16; ++h) {
            const float kv = keb[h] + peb[h];
            const float vv = veb[h] + peb[h];
            ka[c][h] = (_Float16)kv;
            va[c][h] = (_Float16)vv;
            ksp[h]   = kv;
        }
        store8a(kw + k0, ksp);
        store8a(kw + k1, ksp + 8);
    }

    // w logits = k (16x128) @ Mk^T (128x50, padded to 64 cols, masked by mult)
    for (int nt = 0; nt < 4; ++nt) {
        v8f acc = {};
        const int   nrow = nt * 16 + mrow;
        const float msk  = (nrow < SIZE_M) ? 1.0f : 0.0f;
        const float* wr  = Mk + (size_t)(nrow < SIZE_M ? nrow : 0) * DIM_S;
        #pragma unroll
        for (int c = 0; c < 4; ++c) {
            const v16h b = loadB16(wr + c * 32 + g * 8, msk);
            acc = __builtin_amdgcn_wmma_f32_16x16x32_f16(
                      false, ka[c], false, b, (short)0, acc, false, false);
        }
        #pragma unroll
        for (int rr = 0; rr < 8; ++rr)
            lds_logits[wid][g * 8 + rr][nt * 16 + mrow] = acc[rr];
    }
    __syncthreads();

    // softmax over m in [0,50): lanes 0..15 each own one row
    if (lane < 16) {
        const int row  = lane;
        const int ppos = p0 + row;
        float mx = -1e30f;
        for (int m = 0; m < SIZE_M; ++m)
            mx = fmaxf(mx, lds_logits[wid][row][m]);
        float sum = 0.0f;
        for (int m = 0; m < SIZE_M; ++m)
            sum += __expf(lds_logits[wid][row][m] - mx);
        const float inv = 1.0f / sum;
        for (int m = 0; m < SIZE_M; ++m)
            ws_w[(size_t)ppos * 64 + m] =
                __expf(lds_logits[wid][row][m] - mx) * inv;
    }

    // e = sigmoid(v @ e_W^T + e_b), a = tanh(v @ a_W^T + a_b)
    for (int nt = 0; nt < 8; ++nt) {
        v8f ae = {}, aa = {};
        const int nrow = nt * 16 + mrow;
        const float* er = e_W + (size_t)nrow * DIM_S;
        const float* ar = a_W + (size_t)nrow * DIM_S;
        #pragma unroll
        for (int c = 0; c < 4; ++c) {
            const v16h be = loadB16(er + c * 32 + g * 8, 1.0f);
            const v16h ba = loadB16(ar + c * 32 + g * 8, 1.0f);
            ae = __builtin_amdgcn_wmma_f32_16x16x32_f16(
                     false, va[c], false, be, (short)0, ae, false, false);
            aa = __builtin_amdgcn_wmma_f32_16x16x32_f16(
                     false, va[c], false, ba, (short)0, aa, false, false);
        }
        const int   n  = nt * 16 + mrow;
        const float eb = e_b[n];
        const float ab = a_b[n];
        #pragma unroll
        for (int rr = 0; rr < 8; ++rr) {
            const int pp = p0 + g * 8 + rr;
            ws_e[(size_t)pp * DIM_S + n] = 1.0f / (1.0f + __expf(-(ae[rr] + eb)));
            ws_a[(size_t)pp * DIM_S + n] = tanhf(aa[rr] + ab);
        }
    }
}

// ---------------------------------------------------------------------------
// Phase 2: sequential scan over T. One block per batch, thread = column d.
// Mv state (50 rows) lives in registers; read_t needs no cross-lane reduce.
// Streams 329 MB of Mv to d_out -> the HBM-bound part of the problem.
// ---------------------------------------------------------------------------
__global__ void __launch_bounds__(128)
dkvmn_phase2(const float* __restrict__ Mv0, const float* __restrict__ ws_w,
             const float* __restrict__ ws_e, const float* __restrict__ ws_a,
             float* __restrict__ ws_read, float* __restrict__ out_Mv)
{
    __shared__ float wsh[SIZE_M];
    const int b = blockIdx.x;
    const int d = threadIdx.x;          // 0..127

    float Mv[SIZE_M];
    #pragma unroll
    for (int m = 0; m < SIZE_M; ++m) {
        const float v0 = Mv0[m * DIM_S + d];
        Mv[m] = v0;
        out_Mv[(((size_t)b * (TT + 1) + 0) * SIZE_M + m) * DIM_S + d] = v0;
    }

    for (int t = 0; t < TT; ++t) {
        const size_t posi = (size_t)b * TT + t;
        __syncthreads();
        if (threadIdx.x < SIZE_M)
            wsh[threadIdx.x] = ws_w[posi * 64 + threadIdx.x];
        __syncthreads();

        const float e = ws_e[posi * DIM_S + d];
        const float a = ws_a[posi * DIM_S + d];
        float rd = 0.0f;
        float* outp =
            out_Mv + (((size_t)b * (TT + 1) + (t + 1)) * SIZE_M) * DIM_S + d;
        #pragma unroll
        for (int m = 0; m < SIZE_M; ++m) {
            const float w = wsh[m];
            rd += w * Mv[m];                           // read uses pre-update Mv
            Mv[m] = Mv[m] * (1.0f - w * e) + w * a;
            outp[(size_t)m * DIM_S] = Mv[m];
        }
        ws_read[posi * DIM_S + d] = rd;
    }
}

// ---------------------------------------------------------------------------
// Phase 3: f = tanh([read,k] @ f_W^T + f_b) via WMMA (K=256), then
// p = sigmoid(f . p_W + p_b).  Chunks 0..3 come from read, 4..7 from k.
// ---------------------------------------------------------------------------
__global__ void __launch_bounds__(128)
dkvmn_phase3(const float* __restrict__ ws_read, const float* __restrict__ ws_k,
             const float* __restrict__ f_W, const float* __restrict__ f_b,
             const float* __restrict__ p_W, const float* __restrict__ p_b,
             float* __restrict__ out_p)
{
    __shared__ float lds_f[WPB][16][DIM_S];

    const int wid  = threadIdx.x >> 5;
    const int lane = threadIdx.x & 31;
    const int g    = lane >> 4;
    const int mrow = lane & 15;
    const int tile = blockIdx.x * WPB + wid;
    const int p0   = tile * 16;
    const int pos  = p0 + mrow;

    const float* rrow = ws_read + (size_t)pos * DIM_S;
    const float* krow = ws_k   + (size_t)pos * DIM_S;

    v16h af[8];                         // K = 0..255: [read | k]
    #pragma unroll
    for (int c = 0; c < 4; ++c) {
        const int k0 = c * 32 + g * 8;
        const int k1 = c * 32 + 16 + g * 8;
        float t0[16], t1[16];
        load8a(rrow + k0, t0); load8a(rrow + k1, t0 + 8);
        load8a(krow + k0, t1); load8a(krow + k1, t1 + 8);
        #pragma unroll
        for (int h = 0; h < 16; ++h) {
            af[c][h]     = (_Float16)t0[h];
            af[c + 4][h] = (_Float16)t1[h];
        }
    }

    for (int nt = 0; nt < 8; ++nt) {
        v8f acc = {};
        const int nrow = nt * 16 + mrow;
        const float* fr = f_W + (size_t)nrow * (2 * DIM_S);
        #pragma unroll
        for (int c = 0; c < 8; ++c) {
            const v16h bf = loadB16(fr + c * 32 + g * 8, 1.0f);
            acc = __builtin_amdgcn_wmma_f32_16x16x32_f16(
                      false, af[c], false, bf, (short)0, acc, false, false);
        }
        const float fb = f_b[nt * 16 + mrow];
        #pragma unroll
        for (int rr = 0; rr < 8; ++rr)
            lds_f[wid][g * 8 + rr][nt * 16 + mrow] = tanhf(acc[rr] + fb);
    }
    __syncthreads();

    if (lane < 16) {
        const int row = lane;
        float s = p_b[0];
        for (int n = 0; n < DIM_S; ++n)
            s += lds_f[wid][row][n] * p_W[n];
        out_p[p0 + row] = 1.0f / (1.0f + __expf(-s));
    }
}

// ---------------------------------------------------------------------------
extern "C" void kernel_launch(void* const* d_in, const int* in_sizes, int n_in,
                              void* d_out, int out_size, void* d_ws, size_t ws_size,
                              hipStream_t stream) {
    const int*   q       = (const int*)  d_in[0];
    const int*   r       = (const int*)  d_in[1];
    const int*   pid     = (const int*)  d_in[2];
    const float* pid_emb = (const float*)d_in[3];
    const float* k_emb   = (const float*)d_in[4];
    const float* v_emb   = (const float*)d_in[5];
    const float* Mk      = (const float*)d_in[6];
    const float* Mv0     = (const float*)d_in[7];
    const float* f_W     = (const float*)d_in[8];
    const float* f_b     = (const float*)d_in[9];
    const float* p_W     = (const float*)d_in[10];
    const float* p_b     = (const float*)d_in[11];
    const float* e_W     = (const float*)d_in[12];
    const float* e_b     = (const float*)d_in[13];
    const float* a_W     = (const float*)d_in[14];
    const float* a_b     = (const float*)d_in[15];

    float* out_p  = (float*)d_out;          // (B,T) = 12800 floats
    float* out_Mv = out_p + BT;             // (B,T+1,50,128) floats

    // workspace layout (floats): w[BT*64] | e[BT*128] | a[BT*128] | k[BT*128] | read[BT*128]
    float* ws      = (float*)d_ws;
    float* ws_w    = ws;
    float* ws_e    = ws_w + (size_t)BT * 64;
    float* ws_a    = ws_e + (size_t)BT * DIM_S;
    float* ws_k    = ws_a + (size_t)BT * DIM_S;
    float* ws_read = ws_k + (size_t)BT * DIM_S;

    dkvmn_phase1<<<NBLK1, 128, 0, stream>>>(q, r, pid, pid_emb, k_emb, v_emb,
                                            Mk, e_W, e_b, a_W, a_b,
                                            ws_w, ws_e, ws_a, ws_k);
    dkvmn_phase2<<<BB, 128, 0, stream>>>(Mv0, ws_w, ws_e, ws_a, ws_read, out_Mv);
    dkvmn_phase3<<<NBLK1, 128, 0, stream>>>(ws_read, ws_k, f_W, f_b, p_W, p_b,
                                            out_p);
}
